// NeRFSmall_ray_88519275971154
// MI455X (gfx1250) — matrix operand order repaired and checked by reading, not compiled, not run on hardware
//
#include <hip/hip_runtime.h>
#include <math.h>

typedef __attribute__((ext_vector_type(16))) _Float16 v16h;
typedef __attribute__((ext_vector_type(8)))  _Float16 v8h;
typedef __attribute__((ext_vector_type(8)))  float    v8f;

#define N_RAYS   8192
#define N_SAMP   128
#define APP_DIM  32
#define GEO      15

// ---------------------------------------------------------------------------
// WMMA helpers (CDNA5 gfx1250, wave32, V_WMMA_F32_16X16X32_F16)
// ---------------------------------------------------------------------------
__device__ __forceinline__ v8f wmma_f16(v16h a, v16h b, v8f c) {
    // 8 args: (neg_a, A, neg_b, B, c_mod, C, reuse_a, reuse_b)
    return __builtin_amdgcn_wmma_f32_16x16x32_f16(false, a, false, b, (short)0, c,
                                                  false, false);
}

// A-matrix fragment (16x32 f16, M x K): per ISA table:
// lane 0-15 hold row M=lane, K = {0..7, 16..23}; lanes 16-31: K = {8..15, 24..31}
__device__ __forceinline__ v16h load_a_frag(const _Float16* rows, int rowStride,
                                            int row, int kbase, int lane) {
    int kb = kbase + ((lane & 16) ? 8 : 0);
    const _Float16* p = rows + row * rowStride + kb;
    v8h lo = *(const v8h*)(p);        // K = kb+0 .. kb+7
    v8h hi = *(const v8h*)(p + 16);   // K = kb+16 .. kb+23
    return __builtin_shufflevector(lo, hi, 0,1,2,3,4,5,6,7,8,9,10,11,12,13,14,15);
}

// B fragment: pre-swizzled in LDS, contiguous 16 halves per lane (32B aligned)
__device__ __forceinline__ v16h load_b_frag(const _Float16* wf, int frag, int lane) {
    return *(const v16h*)(wf + frag * 512 + lane * 16);
}

// Stage a K x N f32 weight matrix (row-major, stride Nstride) into per-lane
// WMMA B-fragments: frag (kc, nt) covers K-chunk kc*32..+31, N-tile nt*16..+15.
// B-layout (ISA table order): lane l holds column n=l&15;
// halves h -> K = kc*32 + (l>=16 ? 8 : 0) + (h&7) + (h>=8 ? 16 : 0) (matches A).
__device__ void stage_wfrags(_Float16* wf, int fragBase, int nKChunks, int nNTiles,
                             const float* W, int Kreal, int Nreal, int Nstride,
                             int tid, int nthreads) {
    int total = nKChunks * nNTiles * 512;
    for (int idx = tid; idx < total; idx += nthreads) {
        int f      = idx >> 9;
        int within = idx & 511;
        int l      = within >> 4;
        int h      = within & 15;
        int kc = f / nNTiles;
        int nt = f % nNTiles;
        int n = nt * 16 + (l & 15);
        int k = kc * 32 + ((l & 16) ? 8 : 0) + (h & 7) + ((h & 8) ? 16 : 0);
        float v = (k < Kreal && n < Nreal) ? W[k * Nstride + n] : 0.0f;
        wf[(fragBase + f) * 512 + within] = (_Float16)v;
    }
}

// Scatter a D/C fragment (16x16 f32, ISA layout: lane l holds N=l&15,
// M = vgpr + 8*(l>=16)) into f16 LDS rows with fused ReLU.
__device__ __forceinline__ void store_hidden_relu(_Float16* rows, int rowStride,
                                                  int tbase, int lane, int nt, v8f c) {
    int hi8 = (lane & 16) ? 8 : 0;
    int n = nt * 16 + (lane & 15);
    #pragma unroll
    for (int v = 0; v < 8; ++v) {
        float x = c[v];
        x = x > 0.0f ? x : 0.0f;
        rows[(tbase + v + hi8) * rowStride + n] = (_Float16)x;
    }
}

// ---------------------------------------------------------------------------
// Kernel 1: per-point sigma MLP (WMMA) + per-ray alpha compositing.
// One block per ray: 128 threads = 4 waves; each wave does two 16-point tiles.
// ---------------------------------------------------------------------------
__global__ __launch_bounds__(128) void nerf_sigma_composite_kernel(
    const float* __restrict__ pts, const float* __restrict__ fts,
    const float* __restrict__ z_vals, const float* __restrict__ rays_d,
    const float* __restrict__ ws0, const float* __restrict__ ws1,
    const float* __restrict__ ws2,
    float* __restrict__ out_weights, float* __restrict__ out_depth,
    float* __restrict__ out_acc, float* __restrict__ out_featmap)
{
    __shared__ _Float16 wfrag[22 * 512];   // L0:12 frags, L1:8, L2:2 (22.5 KB)
    __shared__ _Float16 rows[128 * 96];    // f16 input / hidden rows (24 KB)
    __shared__ float    rawS[128 * 16];    // raw MLP output per sample (8 KB)
    __shared__ float    scanbuf[128];
    __shared__ float    red[128 * 17];

    const int ray  = blockIdx.x;
    const int tid  = threadIdx.x;
    const int lane = tid & 31;
    const int wave = tid >> 5;

    // ---- stage weights as WMMA B-fragments (f32 -> f16) ----
    stage_wfrags(wfrag,  0, 3, 4, ws0, 95, 64, 64, tid, 128);  // 95x64 (K pad 96)
    stage_wfrags(wfrag, 12, 2, 4, ws1, 64, 64, 64, tid, 128);  // 64x64
    stage_wfrags(wfrag, 20, 2, 1, ws2, 64, 16, 16, tid, 128);  // 64x16

    // ---- stage inputs: thread s builds the 95-d (pad 96) row for sample s ----
    {
        const int s = tid;
        const int p = ray * N_SAMP + s;
        _Float16* r = rows + s * 96;
        const float* fp = fts + (size_t)p * APP_DIM;
        #pragma unroll
        for (int i = 0; i < APP_DIM; ++i) r[i] = (_Float16)fp[i];
        float x = pts[(size_t)p * 3 + 0];
        float y = pts[(size_t)p * 3 + 1];
        float z = pts[(size_t)p * 3 + 2];
        r[32] = (_Float16)x; r[33] = (_Float16)y; r[34] = (_Float16)z;
        #pragma unroll
        for (int i = 0; i < 10; ++i) {
            float f = (float)(1 << i);
            r[35 + 6*i + 0] = (_Float16)__sinf(f * x);
            r[35 + 6*i + 1] = (_Float16)__sinf(f * y);
            r[35 + 6*i + 2] = (_Float16)__sinf(f * z);
            r[35 + 6*i + 3] = (_Float16)__cosf(f * x);
            r[35 + 6*i + 4] = (_Float16)__cosf(f * y);
            r[35 + 6*i + 5] = (_Float16)__cosf(f * z);
        }
        r[95] = (_Float16)0.0f;
    }
    __syncthreads();

    // ---- sigma MLP: each wave runs two 16-sample tiles through WMMA ----
    for (int tl = 0; tl < 2; ++tl) {
        const int tile  = wave * 2 + tl;
        const int tbase = tile * 16;
        const int arow  = tbase + (lane & 15);

        // layer 0: [16x96] x [96x64]
        v8f c0 = {}, c1 = {}, c2 = {}, c3 = {};
        #pragma unroll
        for (int kc = 0; kc < 3; ++kc) {
            v16h a = load_a_frag(rows, 96, arow, kc * 32, lane);
            c0 = wmma_f16(a, load_b_frag(wfrag, kc * 4 + 0, lane), c0);
            c1 = wmma_f16(a, load_b_frag(wfrag, kc * 4 + 1, lane), c1);
            c2 = wmma_f16(a, load_b_frag(wfrag, kc * 4 + 2, lane), c2);
            c3 = wmma_f16(a, load_b_frag(wfrag, kc * 4 + 3, lane), c3);
        }
        __syncthreads();
        store_hidden_relu(rows, 96, tbase, lane, 0, c0);
        store_hidden_relu(rows, 96, tbase, lane, 1, c1);
        store_hidden_relu(rows, 96, tbase, lane, 2, c2);
        store_hidden_relu(rows, 96, tbase, lane, 3, c3);
        __syncthreads();

        // layer 1: [16x64] x [64x64]
        v8f d0 = {}, d1 = {}, d2 = {}, d3 = {};
        #pragma unroll
        for (int kc = 0; kc < 2; ++kc) {
            v16h a = load_a_frag(rows, 96, arow, kc * 32, lane);
            d0 = wmma_f16(a, load_b_frag(wfrag, 12 + kc * 4 + 0, lane), d0);
            d1 = wmma_f16(a, load_b_frag(wfrag, 12 + kc * 4 + 1, lane), d1);
            d2 = wmma_f16(a, load_b_frag(wfrag, 12 + kc * 4 + 2, lane), d2);
            d3 = wmma_f16(a, load_b_frag(wfrag, 12 + kc * 4 + 3, lane), d3);
        }
        __syncthreads();
        store_hidden_relu(rows, 96, tbase, lane, 0, d0);
        store_hidden_relu(rows, 96, tbase, lane, 1, d1);
        store_hidden_relu(rows, 96, tbase, lane, 2, d2);
        store_hidden_relu(rows, 96, tbase, lane, 3, d3);
        __syncthreads();

        // layer 2: [16x64] x [64x16] -> raw (no activation)
        v8f cr = {};
        #pragma unroll
        for (int kc = 0; kc < 2; ++kc) {
            v16h a = load_a_frag(rows, 96, arow, kc * 32, lane);
            cr = wmma_f16(a, load_b_frag(wfrag, 20 + kc, lane), cr);
        }
        {
            int hi8 = (lane & 16) ? 8 : 0;
            int n = lane & 15;
            #pragma unroll
            for (int v = 0; v < 8; ++v)
                rawS[(tbase + v + hi8) * 16 + n] = cr[v];
        }
        __syncthreads();
    }

    // ---- alpha compositing over the 128 samples (thread = sample) ----
    const int s = tid;
    float dx = rays_d[ray * 3 + 0];
    float dy = rays_d[ray * 3 + 1];
    float dz = rays_d[ray * 3 + 2];
    float nrm = sqrtf(dx * dx + dy * dy + dz * dz);
    float zv = z_vals[(size_t)ray * N_SAMP + s];
    float alpha;
    if (s < N_SAMP - 1) {
        float zn   = z_vals[(size_t)ray * N_SAMP + s + 1];
        float dens = fmaxf(rawS[s * 16 + 0], 0.0f);
        alpha = 1.0f - __expf(-dens * (zn - zv) * nrm);
    } else {
        alpha = 1.0f;
    }
    scanbuf[s] = 1.0f + 1e-10f - alpha;
    __syncthreads();
    // Hillis-Steele inclusive product scan
    for (int off = 1; off < N_SAMP; off <<= 1) {
        float other = (s >= off) ? scanbuf[s - off] : 1.0f;
        float mine  = scanbuf[s];
        __syncthreads();
        scanbuf[s] = mine * other;
        __syncthreads();
    }
    float trans = (s == 0) ? 1.0f : scanbuf[s - 1];
    float w = alpha * trans;
    out_weights[(size_t)ray * N_SAMP + s] = w;

    float* rr = red + s * 17;
    rr[0] = w * zv;   // depth contribution
    rr[1] = w;        // acc contribution
    #pragma unroll
    for (int ch = 0; ch < GEO; ++ch)
        rr[2 + ch] = w * fmaxf(rawS[s * 16 + 1 + ch], 0.0f);
    for (int off = 64; off >= 1; off >>= 1) {
        __syncthreads();
        if (s < off) {
            #pragma unroll
            for (int j = 0; j < 17; ++j) red[s * 17 + j] += red[(s + off) * 17 + j];
        }
    }
    __syncthreads();
    if (s == 0) { out_depth[ray] = red[0]; out_acc[ray] = red[1]; }
    if (s < GEO) out_featmap[(size_t)ray * GEO + s] = red[2 + s];
}

// ---------------------------------------------------------------------------
// Kernel 2: per-ray color MLP via WMMA. 256 threads = 8 waves = 128 rays/block.
// ---------------------------------------------------------------------------
__global__ __launch_bounds__(256) void nerf_color_kernel(
    const float* __restrict__ viewdirs,
    const float* __restrict__ wc0, const float* __restrict__ wc1,
    const float* __restrict__ wc2, const float* __restrict__ wc3,
    const float* __restrict__ featmap, float* __restrict__ out_color)
{
    __shared__ _Float16 wfrag[26 * 512];   // 8+8+8+2 fragments (26 KB)
    __shared__ _Float16 rows[128 * 64];    // input / hidden rows (16 KB)

    const int tid  = threadIdx.x;
    const int lane = tid & 31;
    const int wave = tid >> 5;

    stage_wfrags(wfrag,  0, 2, 4, wc0, 42, 64, 64, tid, 256);  // 42x64 (K pad 64)
    stage_wfrags(wfrag,  8, 2, 4, wc1, 64, 64, 64, tid, 256);
    stage_wfrags(wfrag, 16, 2, 4, wc2, 64, 64, 64, tid, 256);
    stage_wfrags(wfrag, 24, 2, 1, wc3, 64,  3,  3, tid, 256);  // 64x3 (N pad 16)

    // stage inputs: thread r (<128) builds ray row: [featmap(15) | dir posenc(27) | 0 pad]
    if (tid < 128) {
        const int r = blockIdx.x * 128 + tid;
        _Float16* row = rows + tid * 64;
        #pragma unroll
        for (int i = 0; i < GEO; ++i) row[i] = (_Float16)featmap[(size_t)r * GEO + i];
        float x = viewdirs[r * 3 + 0];
        float y = viewdirs[r * 3 + 1];
        float z = viewdirs[r * 3 + 2];
        row[15] = (_Float16)x; row[16] = (_Float16)y; row[17] = (_Float16)z;
        #pragma unroll
        for (int i = 0; i < 4; ++i) {
            float f = (float)(1 << i);
            row[18 + 6*i + 0] = (_Float16)__sinf(f * x);
            row[18 + 6*i + 1] = (_Float16)__sinf(f * y);
            row[18 + 6*i + 2] = (_Float16)__sinf(f * z);
            row[21 + 6*i + 0] = (_Float16)__cosf(f * x);
            row[21 + 6*i + 1] = (_Float16)__cosf(f * y);
            row[21 + 6*i + 2] = (_Float16)__cosf(f * z);
        }
        #pragma unroll
        for (int i = 42; i < 64; ++i) row[i] = (_Float16)0.0f;
    }
    __syncthreads();

    const int tbase = wave * 16;
    const int arow  = tbase + (lane & 15);

    // three hidden layers, fragment bases 0 / 8 / 16
    #pragma unroll
    for (int layer = 0; layer < 3; ++layer) {
        const int fb = layer * 8;
        v8f c0 = {}, c1 = {}, c2 = {}, c3 = {};
        #pragma unroll
        for (int kc = 0; kc < 2; ++kc) {
            v16h a = load_a_frag(rows, 64, arow, kc * 32, lane);
            c0 = wmma_f16(a, load_b_frag(wfrag, fb + kc * 4 + 0, lane), c0);
            c1 = wmma_f16(a, load_b_frag(wfrag, fb + kc * 4 + 1, lane), c1);
            c2 = wmma_f16(a, load_b_frag(wfrag, fb + kc * 4 + 2, lane), c2);
            c3 = wmma_f16(a, load_b_frag(wfrag, fb + kc * 4 + 3, lane), c3);
        }
        __syncthreads();
        store_hidden_relu(rows, 64, tbase, lane, 0, c0);
        store_hidden_relu(rows, 64, tbase, lane, 1, c1);
        store_hidden_relu(rows, 64, tbase, lane, 2, c2);
        store_hidden_relu(rows, 64, tbase, lane, 3, c3);
        __syncthreads();
    }

    // output layer: [16x64] x [64x16] (only N<3 valid) -> sigmoid
    v8f cr = {};
    #pragma unroll
    for (int kc = 0; kc < 2; ++kc) {
        v16h a = load_a_frag(rows, 64, arow, kc * 32, lane);
        cr = wmma_f16(a, load_b_frag(wfrag, 24 + kc, lane), cr);
    }
    {
        int n = lane & 15;
        if (n < 3) {
            int hi8 = (lane & 16) ? 8 : 0;
            #pragma unroll
            for (int v = 0; v < 8; ++v) {
                int m = blockIdx.x * 128 + tbase + v + hi8;
                float xv = cr[v];
                out_color[(size_t)m * 3 + n] = 1.0f / (1.0f + __expf(-xv));
            }
        }
    }
}

// ---------------------------------------------------------------------------
// Launch
// ---------------------------------------------------------------------------
extern "C" void kernel_launch(void* const* d_in, const int* in_sizes, int n_in,
                              void* d_out, int out_size, void* d_ws, size_t ws_size,
                              hipStream_t stream) {
    const float* pts      = (const float*)d_in[0];
    const float* viewdirs = (const float*)d_in[1];
    const float* fts      = (const float*)d_in[2];
    const float* z_vals   = (const float*)d_in[3];
    const float* rays_d   = (const float*)d_in[4];
    const float* ws0      = (const float*)d_in[5];
    const float* ws1      = (const float*)d_in[6];
    const float* ws2      = (const float*)d_in[7];
    const float* wc0      = (const float*)d_in[8];
    const float* wc1      = (const float*)d_in[9];
    const float* wc2      = (const float*)d_in[10];
    const float* wc3      = (const float*)d_in[11];

    float* out         = (float*)d_out;
    float* out_color   = out;                          // [8192, 3]
    float* out_depth   = out + (size_t)N_RAYS * 3;     // [8192]
    float* out_acc     = out_depth + N_RAYS;           // [8192]
    float* out_weights = out_acc + N_RAYS;             // [8192, 128]
    float* out_featmap = out_weights + (size_t)N_RAYS * N_SAMP;  // [8192, 15]

    nerf_sigma_composite_kernel<<<N_RAYS, N_SAMP, 0, stream>>>(
        pts, fts, z_vals, rays_d, ws0, ws1, ws2,
        out_weights, out_depth, out_acc, out_featmap);

    nerf_color_kernel<<<N_RAYS / 128, 256, 0, stream>>>(
        viewdirs, wc0, wc1, wc2, wc3, out_featmap, out_color);
}
